// CFConv_5162550690484
// MI455X (gfx1250) — matrix operand order repaired and checked by reading, not compiled, not run on hardware
//
#include <hip/hip_runtime.h>

typedef __attribute__((ext_vector_type(16))) _Float16 v16h;
typedef __attribute__((ext_vector_type(8)))  float    v8f;
typedef __attribute__((ext_vector_type(4)))  float    f4;

#define N_ATOMS   768
#define N_FEAT    128
#define K_RBF     301
#define NS1       10            // K-steps for GEMM1 (301 -> 320, 10 x 32)
#define NS2       4             // K-steps for GEMM2 (128, 4 x 32)
#define GAMMA_F   10.0f
#define SHIFT_F   0.6931471805599453f
#define JSPLIT    4             // waves per output row (j-range split)
#define JTILES    (N_ATOMS / 16 / JSPLIT)   // 12 tiles per wave

// d_ws layout (dwords): W1^T frags | W2^T frags | JSPLIT partial outputs
#define FRAG1_DW  (NS1 * 8 * 32 * 8)   // 20480 dwords = 80 KB
#define FRAG2_DW  (NS2 * 8 * 32 * 8)   //  8192 dwords = 32 KB
#define PART_DW   (N_ATOMS * N_FEAT)   // 98304 dwords per partial
#define PART_OFF  (FRAG1_DW + FRAG2_DW)
// total ws: (28672 + 4*98304)*4 B ~= 1.65 MB

__device__ __forceinline__ float sspf(float xv) {
    // shifted softplus: log(1+exp(x)) - log(2) = log(0.5 + 0.5*exp(x)), guarded
    float lv = __logf(0.5f + 0.5f * __expf(xv));
    return (xv > 20.0f) ? (xv - SHIFT_F) : lv;
}

// Pre-swizzle W1 (301x128, zero-padded to 320) and W2 (128x128) into f16
// WMMA A-fragment layout for A = W^T (M = out-feature tile, K = in dim):
//   dword q = (((s*8 + m)*32 + lane)*8 + v); element e=2v+t maps to
//   k = s*32 + (e<8 ? hi8+e : 16+hi8+(e-8)),  f = 16m + lane%16, hi8=(lane>=16)*8
__global__ void cfconv_pack_weights(const float* __restrict__ W1,
                                    const float* __restrict__ W2,
                                    unsigned* __restrict__ ws) {
    int idx = blockIdx.x * blockDim.x + threadIdx.x;
    if (idx >= FRAG1_DW + FRAG2_DW) return;
    const float* W; int q, kmax;
    if (idx < FRAG1_DW) { W = W1; q = idx;            kmax = K_RBF;  }
    else                { W = W2; q = idx - FRAG1_DW; kmax = N_FEAT; }
    int v    = q & 7;
    int lane = (q >> 3) & 31;
    int m    = (q >> 8) & 7;
    int s    = q >> 11;
    int hi8  = (lane >= 16) ? 8 : 0;
    int f    = 16 * m + (lane & 15);
    union { _Float16 h[2]; unsigned u; } pk;
#pragma unroll
    for (int t = 0; t < 2; ++t) {
        int e = 2 * v + t;
        int k = s * 32 + ((e < 8) ? (hi8 + e) : (16 + hi8 + (e - 8)));
        float val = (k < kmax) ? W[k * N_FEAT + f] : 0.0f;
        pk.h[t] = (_Float16)val;
    }
    ws[idx] = pk.u;
}

// One wave per (output row i, j-range quarter). Per 16-wide j-tile, fully
// fused in registers:
//   rbf^T (B op, v_exp_f32)            -> WMMA -> h^T (C layout, lane = pair)
//   h^T   (B op, packed in-lane from C1) -> WMMA -> w^T -> po += x[j,:]*w
// Biases ride the WMMA accumulators. Fragments are fetched in groups of 4
// (one 8x b128 clause feeding 4 WMMAs) with intra-iteration live ranges only.
__global__ __launch_bounds__(128) __attribute__((amdgpu_waves_per_eu(4)))
void cfconv_main(const float* __restrict__ x,
                 const float* __restrict__ r,
                 const float* __restrict__ b1,
                 const float* __restrict__ b2,
                 const unsigned* __restrict__ ws,
                 float* __restrict__ parts) {
    const int  lane = threadIdx.x & 31;
    const int  wv   = (blockIdx.x * blockDim.x + threadIdx.x) >> 5;
    if (wv >= N_ATOMS * JSPLIT) return;
    const int  i    = wv >> 2;          // output row
    const int  part = wv & (JSPLIT - 1);
    const int  jj   = lane & 15;
    const bool hif  = lane >= 16;
    const int  hi16 = hif ? 16 : 0;
    const int  hi8  = hif ? 8 : 0;

    const float rix = r[3 * i + 0], riy = r[3 * i + 1], riz = r[3 * i + 2];

    const v16h* frag1 = (const v16h*)ws;
    const v16h* frag2 = (const v16h*)ws + (FRAG1_DW / 8);

    float po[8][8];
#pragma unroll
    for (int m = 0; m < 8; ++m)
#pragma unroll
        for (int t = 0; t < 8; ++t) po[m][t] = 0.0f;

    const int j0beg = part * (JTILES * 16);
    for (int jt = 0; jt < JTILES; ++jt) {
        const int j = j0beg + jt * 16 + jj;
        float dx = rix - r[3 * j + 0];
        float dy = riy - r[3 * j + 1];
        float dz = riz - r[3 * j + 2];
        float sq = dx * dx + dy * dy + dz * dz;
        float d  = (sq > 0.0f) ? __builtin_sqrtf(sq) : 0.0f;

        // ---------------- GEMM1: h^T = W1^T x rbf^T + b1 ----------------
        v8f C1[8];
#pragma unroll
        for (int m = 0; m < 8; ++m) {   // accumulator starts at the bias
            f4 bA = *(const f4*)(b1 + 16 * m + hi8);
            f4 bB = *(const f4*)(b1 + 16 * m + hi8 + 4);
#pragma unroll
            for (int t = 0; t < 4; ++t) {
                C1[m][t]     = bA[t];
                C1[m][4 + t] = bB[t];
            }
        }
#pragma unroll 1                        // keep rolled: code stays compact
        for (int s = 0; s < NS1; ++s) {
            v16h bf;
            const int kbase = s * 32 + hi8;   // B layout: element e -> K=kbase+e+(e>=8?8:0)
#pragma unroll
            for (int e = 0; e < 16; ++e) {
                float t = d - 0.1f * (float)(kbase + e + (e >= 8 ? 8 : 0));
                bf[e] = (_Float16)__expf(-GAMMA_F * t * t);  // padded k: exp->0
            }
            const v16h* fr = frag1 + s * (8 * 32) + lane;
#pragma unroll
            for (int g = 0; g < 2; ++g) {  // 4 frags per clause, 4-deep overlap
                v16h a0 = fr[(g * 4 + 0) * 32];
                v16h a1 = fr[(g * 4 + 1) * 32];
                v16h a2 = fr[(g * 4 + 2) * 32];
                v16h a3 = fr[(g * 4 + 3) * 32];
                C1[g * 4 + 0] = __builtin_amdgcn_wmma_f32_16x16x32_f16(
                    false, a0, false, bf, (short)0, C1[g * 4 + 0], false, false);
                C1[g * 4 + 1] = __builtin_amdgcn_wmma_f32_16x16x32_f16(
                    false, a1, false, bf, (short)0, C1[g * 4 + 1], false, false);
                C1[g * 4 + 2] = __builtin_amdgcn_wmma_f32_16x16x32_f16(
                    false, a2, false, bf, (short)0, C1[g * 4 + 2], false, false);
                C1[g * 4 + 3] = __builtin_amdgcn_wmma_f32_16x16x32_f16(
                    false, a3, false, bf, (short)0, C1[g * 4 + 3], false, false);
            }
        }
        // shifted softplus (lane holds h[jj][f = 16m + t + hi8])
#pragma unroll
        for (int m = 0; m < 8; ++m)
#pragma unroll
            for (int t = 0; t < 8; ++t) C1[m][t] = sspf(C1[m][t]);

        // Build all 4 B operands of GEMM2 up front (C1 dies here: packed f16
        // fragments need only 32 VGPRs vs 64 for C1).
        v16h bf2s[NS2];
#pragma unroll
        for (int s2 = 0; s2 < NS2; ++s2) {
#pragma unroll
            for (int rr = 0; rr < 8; ++rr) {
                float a0  = C1[2 * s2][rr];       // block 2s2  : f-off rr+hi8
                float a1  = C1[2 * s2 + 1][rr];   // block 2s2+1: f-off rr+hi8
                float sa0 = __shfl_xor(a0, 16, 32);
                float sa1 = __shfl_xor(a1, 16, 32);
                bf2s[s2][rr]     = (_Float16)a0;  // k = hi8+rr
                bf2s[s2][8 + rr] = (_Float16)a1;  // k = 16+hi8+rr
            }
        }

        // ---------------- GEMM2: w^T = W2^T x h^T + b2 (m-outer) ----------
        const float* xrow = x + j * N_FEAT;
#pragma unroll
        for (int m = 0; m < 8; ++m) {
            // all 4 fragments for this m up front (one clause, 4-deep overlap)
            v16h g0 = frag2[(0 * 8 + m) * 32 + lane];
            v16h g1 = frag2[(1 * 8 + m) * 32 + lane];
            v16h g2 = frag2[(2 * 8 + m) * 32 + lane];
            v16h g3 = frag2[(3 * 8 + m) * 32 + lane];
            f4 bA = *(const f4*)(b2 + 16 * m + hi8);
            f4 bB = *(const f4*)(b2 + 16 * m + hi8 + 4);
            v8f C2;
#pragma unroll
            for (int t = 0; t < 4; ++t) {   // accumulator starts at the bias
                C2[t]     = bA[t];
                C2[4 + t] = bB[t];
            }
            C2 = __builtin_amdgcn_wmma_f32_16x16x32_f16(
                false, g0, false, bf2s[0], (short)0, C2, false, false);
            C2 = __builtin_amdgcn_wmma_f32_16x16x32_f16(
                false, g1, false, bf2s[1], (short)0, C2, false, false);
            C2 = __builtin_amdgcn_wmma_f32_16x16x32_f16(
                false, g2, false, bf2s[2], (short)0, C2, false, false);
            C2 = __builtin_amdgcn_wmma_f32_16x16x32_f16(
                false, g3, false, bf2s[3], (short)0, C2, false, false);

            // ssp + fused o-accumulation: po += x[j, f] * w[jj, f]
            f4 xA = *(const f4*)(xrow + 16 * m + hi8);
            f4 xB = *(const f4*)(xrow + 16 * m + hi8 + 4);
#pragma unroll
            for (int t = 0; t < 4; ++t) {
                po[m][t]     += xA[t] * sspf(C2[t]);
                po[m][4 + t] += xB[t] * sspf(C2[4 + t]);
            }
        }
    }

    // reduce the 16 pairs held across lanes of each half-wave
#pragma unroll
    for (int m = 0; m < 8; ++m)
#pragma unroll
        for (int t = 0; t < 8; ++t) {
            float v = po[m][t];
            v += __shfl_xor(v, 1, 32);
            v += __shfl_xor(v, 2, 32);
            v += __shfl_xor(v, 4, 32);
            v += __shfl_xor(v, 8, 32);
            po[m][t] = v;
        }
    if (jj == 0) {  // lane 0 writes f = 16m+0..7, lane 16 writes f = 16m+8..15
        float* prow = parts + part * PART_DW + i * N_FEAT;
#pragma unroll
        for (int m = 0; m < 8; ++m) {
            f4 a = { po[m][0], po[m][1], po[m][2], po[m][3] };
            f4 b = { po[m][4], po[m][5], po[m][6], po[m][7] };
            *(f4*)(prow + 16 * m + hi8)     = a;
            *(f4*)(prow + 16 * m + hi8 + 4) = b;
        }
    }
}

// Deterministic fixed-order reduction of the JSPLIT partials (no FP atomics).
__global__ void cfconv_reduce(const float* __restrict__ parts,
                              float* __restrict__ out) {
    int idx = blockIdx.x * blockDim.x + threadIdx.x;
    if (idx >= PART_DW) return;
    float s = parts[idx];
#pragma unroll
    for (int p = 1; p < JSPLIT; ++p) s += parts[p * PART_DW + idx];
    out[idx] = s;
}

extern "C" void kernel_launch(void* const* d_in, const int* in_sizes, int n_in,
                              void* d_out, int out_size, void* d_ws, size_t ws_size,
                              hipStream_t stream) {
    const float* x  = (const float*)d_in[0];
    const float* r  = (const float*)d_in[1];
    const float* W1 = (const float*)d_in[2];
    const float* b1 = (const float*)d_in[3];
    const float* W2 = (const float*)d_in[4];
    const float* b2 = (const float*)d_in[5];
    float*    out = (float*)d_out;
    unsigned* ws  = (unsigned*)d_ws;   // needs ~1.65 MB of scratch
    float*  parts = (float*)ws + PART_OFF;

    const int packN = FRAG1_DW + FRAG2_DW;
    cfconv_pack_weights<<<(packN + 255) / 256, 256, 0, stream>>>(W1, W2, ws);

    const int waves = N_ATOMS * JSPLIT;             // 3072 waves
    cfconv_main<<<(waves * 32) / 128, 128, 0, stream>>>(x, r, b1, b2, ws, parts);

    cfconv_reduce<<<(PART_DW + 255) / 256, 256, 0, stream>>>(parts, out);

    (void)in_sizes; (void)n_in; (void)out_size; (void)ws_size;
}
